// self_attn1_19344532701358
// MI455X (gfx1250) — matrix operand order repaired and checked by reading, not compiled, not run on hardware
//
#include <hip/hip_runtime.h>
#include <hip/hip_bf16.h>

// ---------------- problem dims ----------------
#define BB 4
#define CC 256
#define NN 4096
#define NT 256          // NN/16 : 16-wide i/j tiles per batch

// ---------------- workspace byte offsets ----------------
#define WQA_OFF 0u          // wq packed A-form : 2*8*32*16 bf16 = 16 KB
#define WKA_OFF 16384u      // wk packed A-form : 16 KB
#define WVB_OFF 32768u      // wv packed B-form : 16*8*32*16 bf16 = 128 KB
#define QBF_OFF 163840u     // q  plain bf16 [B][32][N]       = 1 MB
#define KPK_OFF 1212416u    // k  swizzled A-tiles [B][NT][32][16] = 1 MB
#define VPK_OFF 2260992u    // v  swizzled A-tiles [B][16][128][32][16] = 8 MB

// async global->LDS path (CDNA5), guarded so the file always compiles
#if defined(__has_builtin)
#if __has_builtin(__builtin_amdgcn_global_load_async_to_lds_b128) && \
    __has_builtin(__builtin_amdgcn_s_wait_asynccnt)
#define HAVE_ASYNC 1
#endif
#endif
#ifndef HAVE_ASYNC
#define HAVE_ASYNC 0
#endif

typedef __attribute__((ext_vector_type(16))) __bf16 v16bf;
typedef __attribute__((ext_vector_type(8)))  float  v8f;
typedef int v4i __attribute__((__vector_size__(4 * sizeof(int))));
typedef __attribute__((address_space(1))) v4i glb_v4i;
typedef __attribute__((address_space(3))) v4i lds_v4i;

__device__ __forceinline__ unsigned short f2bf(float f) {
    unsigned u = __float_as_uint(f);
    unsigned r = u + 0x7FFFu + ((u >> 16) & 1u);   // round-to-nearest-even
    return (unsigned short)(r >> 16);
}

struct BfTile {
    union { v16bf v; unsigned short u[16]; uint4 q[2]; };
};

__device__ __forceinline__ BfTile ld_tile(const unsigned short* p) {
    BfTile t;
    t.q[0] = *(const uint4*)p;
    t.q[1] = *(const uint4*)(p + 8);
    return t;
}
__device__ __forceinline__ BfTile ld_tile2(const unsigned short* p0,
                                           const unsigned short* p1) {
    BfTile t;
    t.q[0] = *(const uint4*)p0;
    t.q[1] = *(const uint4*)p1;
    return t;
}
__device__ __forceinline__ v8f wmma_bf(const BfTile& a, const BfTile& b, v8f c) {
    return __builtin_amdgcn_wmma_f32_16x16x32_bf16(false, a.v, false, b.v,
                                                   (short)0, c, false, false);
}

// ---------------- weight pack kernels ----------------
// A-form: lane l holds row rtile*16+(l&15); halves h -> k = kc*32 + (l<16?0:8) + (h<8?h:h+8)
__global__ void pack_w_a(const float* __restrict__ w, unsigned short* __restrict__ out,
                         int units) {
    int t = blockIdx.x * blockDim.x + threadIdx.x;
    int unit = t >> 5, lane = t & 31;
    if (unit >= units) return;
    int rtile = unit >> 3, kc = unit & 7;
    int r  = rtile * 16 + (lane & 15);
    int kb = kc * 32 + ((lane & 16) ? 8 : 0);
    unsigned short* dst = out + (size_t)t * 16;
#pragma unroll
    for (int h = 0; h < 16; ++h) {
        int k = kb + (h < 8 ? h : h + 8);
        dst[h] = f2bf(w[r * 256 + k]);
    }
}
// B-form: lane l holds col ctile*16+(l&15); halves h -> k = kc*32 + (l<16?0:16) + h
__global__ void pack_w_b(const float* __restrict__ w, unsigned short* __restrict__ out,
                         int units) {
    int t = blockIdx.x * blockDim.x + threadIdx.x;
    int unit = t >> 5, lane = t & 31;
    if (unit >= units) return;
    int ctile = unit >> 3, kc = unit & 7;
    int c  = ctile * 16 + (lane & 15);
    int kb = kc * 32 + ((lane & 16) ? 16 : 0);
    unsigned short* dst = out + (size_t)t * 16;
#pragma unroll
    for (int h = 0; h < 16; ++h)
        dst[h] = f2bf(w[c * 256 + kb + h]);
}

// ---------------- projection kernel (q/k/v via WMMA) ----------------
#define XSTR 264   // halves per LDS row (256 + 8 pad); 528 B, 16B aligned, odd 16B-chunk stride
__global__ __launch_bounds__(256) void proj_kernel(
    const float* __restrict__ x,
    const float* __restrict__ bq, const float* __restrict__ bk,
    const float* __restrict__ bv,
    const unsigned short* __restrict__ wqa, const unsigned short* __restrict__ wka,
    const unsigned short* __restrict__ wvb,
    unsigned short* __restrict__ qbf, unsigned short* __restrict__ kpk,
    unsigned short* __restrict__ vpk) {
    __shared__ unsigned short xs[64 * XSTR];   // bf16 X tile [i=64][k=256]
    int b = blockIdx.x >> 6;
    int iblock = blockIdx.x & 63;
    int i0 = iblock * 64;
    int tid = threadIdx.x;

    {   // stage & convert X tile (coalesced over i)
        int i  = tid & 63;
        int ks = tid >> 6;                       // 0..3 -> k ranges of 64
        const float* xp = x + ((size_t)b * CC + ks * 64) * NN + i0 + i;
        unsigned short* xrow = xs + i * XSTR + ks * 64;
#pragma unroll 4
        for (int kk = 0; kk < 64; ++kk)
            xrow[kk] = f2bf(xp[(size_t)kk * NN]);
    }
    __syncthreads();

    int w = tid >> 5, lane = tid & 31;
    int l15 = lane & 15;
    bool lo = lane < 16;
    const v8f vz = {0.f,0.f,0.f,0.f,0.f,0.f,0.f,0.f};

    for (int u = w; u < 40; u += 8) {
        if (u < 8) {
            // ---- Q (u<4) / K (u<8) : D[o, i] = W * X ----
            bool isQ = u < 4;
            int itile = isQ ? u : u - 4;
            const unsigned short* wp = isQ ? wqa : wka;
            const unsigned short* xrow = xs + (itile * 16 + l15) * XSTR;
            int boff = lo ? 0 : 8;               // B-operand K half select
            v8f a0 = vz, a1 = vz;
#pragma unroll
            for (int kc = 0; kc < 8; ++kc) {
                BfTile A0 = ld_tile(wp + (size_t)(kc * 32 + lane) * 16);
                BfTile A1 = ld_tile(wp + (size_t)((8 + kc) * 32 + lane) * 16);
                BfTile Bx = ld_tile2(xrow + kc * 32 + boff, xrow + kc * 32 + 16 + boff);
                a0 = wmma_bf(A0, Bx, a0);
                a1 = wmma_bf(A1, Bx, a1);
            }
            const float* bias = isQ ? bq : bk;
            int cb = lo ? 0 : 8;
#pragma unroll
            for (int r = 0; r < 8; ++r) { a0[r] += bias[cb + r]; a1[r] += bias[16 + cb + r]; }
            int itile_g = iblock * 4 + itile;
            if (isQ) {  // plain [B][32][N] store (B-operand rebuilt at attention time)
                int ig = itile_g * 16 + l15;
#pragma unroll
                for (int r = 0; r < 8; ++r) {
                    qbf[(size_t)(b * 32 + cb + r) * NN + ig]      = f2bf(a0[r]);
                    qbf[(size_t)(b * 32 + 16 + cb + r) * NN + ig] = f2bf(a1[r]);
                }
            } else {    // lane-local pack into GEMM1 A-layout
                unsigned short* dst = kpk + (size_t)((b * NT + itile_g) * 32 + lane) * 16;
                BfTile t;
#pragma unroll
                for (int h = 0; h < 8; ++h) { t.u[h] = f2bf(a0[h]); t.u[8 + h] = f2bf(a1[h]); }
                *(uint4*)dst = t.q[0]; *(uint4*)(dst + 8) = t.q[1];
            }
        } else {
            // ---- V transposed: D[j, c] = X^T * Wv^T ----
            int vu = u - 8;
            int ct = vu & 15, jp = vu >> 4;      // jp 0..1 -> j-pair of itiles
            const unsigned short* xrow0 = xs + ((2 * jp + 0) * 16 + l15) * XSTR;
            const unsigned short* xrow1 = xs + ((2 * jp + 1) * 16 + l15) * XSTR;
            int aoff = lo ? 0 : 16;              // A-operand K half select
            v8f a0 = vz, a1 = vz;
#pragma unroll
            for (int kc = 0; kc < 8; ++kc) {
                BfTile Bw = ld_tile(wvb + (size_t)((ct * 8 + kc) * 32 + lane) * 16);
                BfTile A0 = ld_tile2(xrow0 + kc * 32 + aoff, xrow0 + kc * 32 + 8 + aoff);
                BfTile A1 = ld_tile2(xrow1 + kc * 32 + aoff, xrow1 + kc * 32 + 8 + aoff);
                a0 = wmma_bf(A0, Bw, a0);
                a1 = wmma_bf(A1, Bw, a1);
            }
            float bvv = bv[ct * 16 + l15];
#pragma unroll
            for (int r = 0; r < 8; ++r) { a0[r] += bvv; a1[r] += bvv; }
            int jbg = iblock * 2 + jp;
            unsigned short* dst =
                vpk + (size_t)(((b * 16 + ct) * 128 + jbg) * 32 + lane) * 16;
            BfTile t;
#pragma unroll
            for (int h = 0; h < 8; ++h) { t.u[h] = f2bf(a0[h]); t.u[8 + h] = f2bf(a1[h]); }
            *(uint4*)dst = t.q[0]; *(uint4*)(dst + 8) = t.q[1];
        }
    }
}

// ---------------- fused sigmoid-attention kernel ----------------
// Each wave owns one 16-wide i-tile and all 16 c-tiles (128 f32 accum VGPRs).
// K/V j-slices are staged once per WG into LDS through the CDNA5 async
// global->LDS path (ASYNCcnt), triple-buffered, removing 8x duplicated L2 reads.
__global__ __launch_bounds__(256) void attn_kernel(
    const float* __restrict__ x, const unsigned short* __restrict__ qbf,
    const unsigned short* __restrict__ kpk, const unsigned short* __restrict__ vpk,
    const float* __restrict__ gamma, float* __restrict__ out) {
    int b = blockIdx.x >> 5;
    int iblock = blockIdx.x & 31;
    int w = threadIdx.x >> 5, lane = threadIdx.x & 31;
    int l15 = lane & 15;
    bool lo = lane < 16;
    int itile_g = iblock * 8 + w;

    // Q as GEMM1 B-operand: lane = i column, halves = c (0..7,16..23 lo / 8..15,24..31 hi)
    BfTile qb;
    {
        int ig = itile_g * 16 + l15;
        int cb = lo ? 0 : 8;
#pragma unroll
        for (int h = 0; h < 16; ++h)
            qb.u[h] = qbf[(size_t)(b * 32 + cb + (h < 8 ? h : h + 8)) * NN + ig];
    }

    const v8f vz = {0.f,0.f,0.f,0.f,0.f,0.f,0.f,0.f};
    v8f acc[16];
#pragma unroll
    for (int ct = 0; ct < 16; ++ct) acc[ct] = vz;

#if HAVE_ASYNC
    // 3 buffers x 18 tiles (0,1 = K jt0/jt1 ; 2..17 = V ct0..15) x 1 KB
    __shared__ unsigned short stg[3 * 18 * 512];
    auto issue_tile = [&](int t, int n, int bufo) {
        const unsigned short* src;
        if (t < 2) src = kpk + (size_t)((b * NT + 2 * n + t) * 32) * 16 + lane * 16;
        else       src = vpk + (size_t)(((b * 16 + (t - 2)) * 128 + n) * 32) * 16 + lane * 16;
        unsigned short* dst = stg + bufo + t * 512 + lane * 16;
        glb_v4i* gsrc = (glb_v4i*)(unsigned long long)src;
        lds_v4i* ldst = (lds_v4i*)(unsigned int)(unsigned long long)dst;
        __builtin_amdgcn_global_load_async_to_lds_b128(gsrc, ldst, 0, 0);
        __builtin_amdgcn_global_load_async_to_lds_b128(gsrc + 1, ldst + 1, 0, 0);
    };
    auto issue_batch = [&](int n) {
        if (n >= 128) return;
        int bufo = (n % 3) * (18 * 512);
        issue_tile(w, n, bufo);
        issue_tile(w + 8, n, bufo);
        if (w < 2) issue_tile(16 + w, n, bufo);
    };
    issue_batch(0);
    issue_batch(1);
    int buf = 0;
#else
    const unsigned short* kbase = kpk + ((size_t)b * NT * 32 + lane) * 16;
    const unsigned short* vbase =
        vpk + (size_t)b * (16u * 128u * 32u * 16u) + (size_t)lane * 16;
#endif

    for (int jb = 0; jb < 128; ++jb) {
#if HAVE_ASYNC
        // oldest outstanding batch (= jb) complete; per-wave in-flight counts
        if (jb >= 126)      __builtin_amdgcn_s_wait_asynccnt(0);
        else if (w < 2)     __builtin_amdgcn_s_wait_asynccnt(6);
        else                __builtin_amdgcn_s_wait_asynccnt(4);
        __syncthreads();          // slices visible; previous iteration's buffer free to overwrite
        issue_batch(jb + 2);
        const unsigned short* sb = stg + buf * (18 * 512);
        BfTile ka0 = ld_tile(sb + lane * 16);
        BfTile ka1 = ld_tile(sb + 512 + lane * 16);
#else
        BfTile ka0 = ld_tile(kbase + (size_t)(2 * jb) * 512);
        BfTile ka1 = ld_tile(kbase + (size_t)(2 * jb) * 512 + 512);
#endif
        // S' = K · Q^T  (j rows, i cols), two 16-row j tiles
        v8f s0 = wmma_bf(ka0, qb, vz);
        v8f s1 = wmma_bf(ka1, qb, vz);

        // sigmoid + half-lane exchange -> GEMM2 B-layout (K=j in lane halves)
        float sv0[8], sv1[8], y0[8], y1[8];
#pragma unroll
        for (int r = 0; r < 8; ++r) {
            sv0[r] = 1.0f / (1.0f + __expf(-s0[r]));
            sv1[r] = 1.0f / (1.0f + __expf(-s1[r]));
        }
#pragma unroll
        for (int r = 0; r < 8; ++r) {
            y0[r] = __shfl_xor(sv0[r], 16, 32);
            y1[r] = __shfl_xor(sv1[r], 16, 32);
        }
        BfTile pb;
#pragma unroll
        for (int h = 0; h < 8; ++h) {
            pb.u[h]     = f2bf(sv0[h]);   // K = j 0..7  / 8..15
            pb.u[8 + h] = f2bf(sv1[h]);   // K = j 16..23 / 24..31
        }

        // O[c,i] += V[c, j0..j0+31] * sigmoid(S)^T  : one WMMA per c-tile
#if HAVE_ASYNC
#pragma unroll
        for (int ct = 0; ct < 16; ++ct) {
            BfTile va = ld_tile(sb + (2 + ct) * 512 + lane * 16);
            acc[ct] = wmma_bf(va, pb, acc[ct]);
        }
        buf = (buf == 2) ? 0 : buf + 1;
#else
        const unsigned short* vj = vbase + (size_t)jb * 512;
#pragma unroll
        for (int ct = 0; ct < 16; ++ct) {
            BfTile va = ld_tile(vj + (size_t)ct * 65536);
            acc[ct] = wmma_bf(va, pb, acc[ct]);
        }
#endif
    }

    // epilogue: out = gamma * O + x
    float g = gamma[0];
    int ig = itile_g * 16 + l15;
    int cb = lo ? 0 : 8;
#pragma unroll
    for (int ct = 0; ct < 16; ++ct) {
#pragma unroll
        for (int r = 0; r < 8; ++r) {
            size_t idx = (size_t)(b * CC + ct * 16 + cb + r) * NN + ig;
            out[idx] = g * acc[ct][r] + x[idx];
        }
    }
}

extern "C" void kernel_launch(void* const* d_in, const int* in_sizes, int n_in,
                              void* d_out, int out_size, void* d_ws, size_t ws_size,
                              hipStream_t stream) {
    (void)in_sizes; (void)n_in; (void)out_size; (void)ws_size;
    const float* x     = (const float*)d_in[0];
    const float* wq    = (const float*)d_in[1];
    const float* bq    = (const float*)d_in[2];
    const float* wk    = (const float*)d_in[3];
    const float* bk    = (const float*)d_in[4];
    const float* wv    = (const float*)d_in[5];
    const float* bv    = (const float*)d_in[6];
    const float* gamma = (const float*)d_in[7];

    char* ws = (char*)d_ws;
    unsigned short* wqa = (unsigned short*)(ws + WQA_OFF);
    unsigned short* wka = (unsigned short*)(ws + WKA_OFF);
    unsigned short* wvb = (unsigned short*)(ws + WVB_OFF);
    unsigned short* qbf = (unsigned short*)(ws + QBF_OFF);
    unsigned short* kpk = (unsigned short*)(ws + KPK_OFF);
    unsigned short* vpk = (unsigned short*)(ws + VPK_OFF);

    pack_w_a<<<2, 256, 0, stream>>>(wq, wqa, 16);
    pack_w_a<<<2, 256, 0, stream>>>(wk, wka, 16);
    pack_w_b<<<16, 256, 0, stream>>>(wv, wvb, 128);
    proj_kernel<<<256, 256, 0, stream>>>(x, bq, bk, bv, wqa, wka, wvb, qbf, kpk, vpk);
    attn_kernel<<<128, 256, 0, stream>>>(x, qbf, kpk, vpk, gamma, (float*)d_out);
}
